// _NoisyTopkRouter_19267223290599
// MI455X (gfx1250) — matrix operand order repaired and checked by reading, not compiled, not run on hardware
//
#include <hip/hip_runtime.h>
#include <stdint.h>

// ---------------------------------------------------------------------------
// NoisyTopkRouter fused kernel for gfx1250 (MI455X).
//   logits       = A @ W_route + b_route
//   noise_logits = A @ W_noise + b_noise
//   noisy        = logits + noise_u * softplus(noise_logits)
//   top-2 over E=64, masked softmax, return (probs, indices)
// A: [N=32768, D=1024] f32.  Weights: [D, E=64] f32 -> bf16, transposed AND
// route/noise interleaved per 16-wide K-chunk so one 64B load per lane feeds
// two back-to-back WMMAs (route + noise share the A fragment).
// GEMMs via v_wmma_f32_16x16x32_bf16 (fp32 accumulate), epilogue fused.
// Schedule forced with sched_group_barrier: all VMEM reads -> pack VALU ->
// 8 WMMAs, so B-fragment waits become partial instead of 8 full stalls.
// A-stream prefetched 2 K-iterations deep with non-temporal hints.
// ---------------------------------------------------------------------------

typedef __attribute__((ext_vector_type(32))) __bf16 v32bf;
typedef __attribute__((ext_vector_type(16))) __bf16 v16bf;
typedef __attribute__((ext_vector_type(8)))  float  v8f;
typedef __attribute__((ext_vector_type(4)))  float  v4f;

#define D_DIM 1024
#define E_DIM 64
#define N_ROWS (8 * 4096)

__device__ __forceinline__ uint32_t bf16_rne(uint32_t u) {
  return (u + 0x7FFFu + ((u >> 16) & 1u)) >> 16;
}

__device__ __forceinline__ uint32_t pack_bf16x2(float lo, float hi) {
  uint32_t ul = bf16_rne(__float_as_uint(lo));
  uint32_t uh = bf16_rne(__float_as_uint(hi));
  return ul | (uh << 16);
}

// Combined weight layout in d_ws (bf16 bits, 256KB total):
//   WtC[n][c][s][t],  n = expert row (64), c = K-chunk of 16 (64),
//                     s = 0:route 1:noise, t = 0..15 within chunk.
// uint16 index = n*2048 + c*32 + s*16 + t.  One lane's fragment-pair for
// (n, kB=c*16) is the 64B at n*2048 + c*32: route frag then noise frag.
__global__ void _router_prep_weights(const float* __restrict__ Wr,
                                     const float* __restrict__ Wn,
                                     uint16_t* __restrict__ WtC) {
  const int e = blockIdx.x;  // 0..63
  for (int d = threadIdx.x; d < D_DIM; d += blockDim.x) {
    const uint32_t r = bf16_rne(__float_as_uint(Wr[(size_t)d * E_DIM + e]));
    const uint32_t n = bf16_rne(__float_as_uint(Wn[(size_t)d * E_DIM + e]));
    const int base = e * 2048 + (d >> 4) * 32 + (d & 15);
    WtC[base]      = (uint16_t)r;
    WtC[base + 16] = (uint16_t)n;
  }
}

__device__ __forceinline__ void load_a_tile(const float* __restrict__ p, v4f t[4]) {
  t[0] = __builtin_nontemporal_load((const v4f*)(p));
  t[1] = __builtin_nontemporal_load((const v4f*)(p + 4));
  t[2] = __builtin_nontemporal_load((const v4f*)(p + 8));
  t[3] = __builtin_nontemporal_load((const v4f*)(p + 12));
}

union WPair { v32bf w; v16bf h[2]; };

// sched_group_barrier masks (LLVM AMDGPU): 0x2=VALU, 0x8=MFMA/WMMA,
// 0x20=VMEM read.
__device__ __forceinline__ void force_k_schedule() {
#if __has_builtin(__builtin_amdgcn_sched_group_barrier)
  __builtin_amdgcn_sched_group_barrier(0x020, 20, 0);  // all VMEM reads first
  __builtin_amdgcn_sched_group_barrier(0x002, 48, 0);  // pack / pipeline VALU
  __builtin_amdgcn_sched_group_barrier(0x008, 8, 0);   // then the 8 WMMAs
#endif
}

__global__ __launch_bounds__(128, 1) void _router_main(
    const float*    __restrict__ A,      // [N, D]
    const float*    __restrict__ bR,     // [E]
    const float*    __restrict__ bN,     // [E]
    const float*    __restrict__ NU,     // [N, E]
    const uint16_t* __restrict__ WtC,    // combined bf16 weights
    float*          __restrict__ out,    // [N, E]
    int*            __restrict__ idx_out)// [N, 2]
{
  const int lane = threadIdx.x & 31;
  const int wave = threadIdx.x >> 5;       // 4 waves / block
  const int lid  = lane & 15;
  const int half = lane >> 4;
  const int rowBase = blockIdx.x * 64 + wave * 16;  // 16 rows per wave

  v8f accR[4], accN[4];
#pragma unroll
  for (int nt = 0; nt < 4; ++nt) { accR[nt] = (v8f)0.0f; accN[nt] = (v8f)0.0f; }

  // A fragment k order used here (contiguous, same as B): lanes 0-15 row M=lane, K=0-15;
  // lanes 16-31 row M=lane-16, K=16-31.
  const float* aRow  = A + (size_t)(rowBase + lid) * D_DIM;
  const int    k0off = 16 * half;

  // B fragment-pair base pointers: lane handles expert row n = nt*16 + lid;
  // lanes 0-15 take K-chunk c = kk/16, lanes 16-31 take c = kk/16 + 1
  // (K=16-31 of the 32-wide step). uint16 offset for step kk is kk*2.
  const uint16_t* pC[4];
#pragma unroll
  for (int nt = 0; nt < 4; ++nt) {
    const int n = nt * 16 + lid;
    pC[nt] = WtC + (size_t)n * 2048 + half * 32;
  }

  // Prologue: 2-deep A pipeline (tiles for kk=0 and kk=32 in flight).
  v4f a0[4], a1[4];
  load_a_tile(aRow + k0off, a0);
  load_a_tile(aRow + 32 + k0off, a1);

  for (int kk = 0; kk < D_DIM; kk += 32) {
    // 1) Issue ALL B fragment-pair loads (WGP$/L2-resident weights). The
    //    sched_group_barrier below keeps them ahead of the WMMAs so the
    //    per-WMMA waits are partial counter decrements, not full stalls.
    WPair P[4];
#pragma unroll
    for (int nt = 0; nt < 4; ++nt) {
      P[nt].w = *(const v32bf*)(pC[nt] + (size_t)kk * 2);
    }

    // 2) Pack the oldest A tile fp32 -> bf16 (RNE); loaded 2 iterations ago.
    union { v16bf v; uint32_t u[8]; } af;
    af.u[0] = pack_bf16x2(a0[0].x, a0[0].y); af.u[1] = pack_bf16x2(a0[0].z, a0[0].w);
    af.u[2] = pack_bf16x2(a0[1].x, a0[1].y); af.u[3] = pack_bf16x2(a0[1].z, a0[1].w);
    af.u[4] = pack_bf16x2(a0[2].x, a0[2].y); af.u[5] = pack_bf16x2(a0[2].z, a0[2].w);
    af.u[6] = pack_bf16x2(a0[3].x, a0[3].y); af.u[7] = pack_bf16x2(a0[3].z, a0[3].w);

    // 3) Rotate the A pipeline and issue loads for kk+64 (uniform wrap keeps
    //    the address valid, no divergence); in flight through this iteration
    //    and the next.
#pragma unroll
    for (int t = 0; t < 4; ++t) a0[t] = a1[t];
    int kn = kk + 64;
    kn = (kn < D_DIM) ? kn : (kn - D_DIM);
    load_a_tile(aRow + kn + k0off, a1);

    // 4) 8 WMMAs sharing one A fragment (route + noise x 4 expert tiles).
#pragma unroll
    for (int nt = 0; nt < 4; ++nt) {
      accR[nt] = __builtin_amdgcn_wmma_f32_16x16x32_bf16(
          false, af.v, false, P[nt].h[0], (short)0, accR[nt], false, false);
      accN[nt] = __builtin_amdgcn_wmma_f32_16x16x32_bf16(
          false, af.v, false, P[nt].h[1], (short)0, accN[nt], false, false);
    }

    force_k_schedule();
  }

  // Biases for this lane's 4 expert columns (fp32).
  float bRv[4], bNv[4];
#pragma unroll
  for (int nt = 0; nt < 4; ++nt) {
    bRv[nt] = bR[nt * 16 + lid];
    bNv[nt] = bN[nt * 16 + lid];
  }

  const float NEG_INF = -__builtin_inff();

  // C/D layout: VGPR j holds row M=j (lanes 0-15) and M=j+8 (lanes 16-31),
  // column N = lid. The two halves reduce independent rows in parallel.
#pragma unroll
  for (int j = 0; j < 8; ++j) {
    const int row = rowBase + j + 8 * half;

    float v1 = NEG_INF, v2 = NEG_INF;
    int   i1 = 0x7FFFFFFF, i2 = 0x7FFFFFFF;

#pragma unroll
    for (int nt = 0; nt < 4; ++nt) {
      const int e = nt * 16 + lid;
      float nl = accN[nt][j] + bNv[nt];
      // numerically-stable softplus
      float sp = fmaxf(nl, 0.0f) + __logf(1.0f + __expf(-fabsf(nl)));
      float nu = __builtin_nontemporal_load(NU + (size_t)row * E_DIM + e);
      float x  = accR[nt][j] + bRv[nt] + nu * sp;
      accR[nt][j] = x;  // keep noisy logit for the write phase
      // lane-local top-2 insert (ties -> lowest index, matching lax.top_k)
      if (x > v1 || (x == v1 && e < i1)) { v2 = v1; i2 = i1; v1 = x; i1 = e; }
      else if (x > v2 || (x == v2 && e < i2)) { v2 = x; i2 = e; }
    }

    // Cross-lane top-2 merge within each 16-lane half (xor 1,2,4,8 stays in half).
#pragma unroll
    for (int m = 1; m < 16; m <<= 1) {
      float w1 = __shfl_xor(v1, m, 32); int j1 = __shfl_xor(i1, m, 32);
      float w2 = __shfl_xor(v2, m, 32); int j2 = __shfl_xor(i2, m, 32);
      bool  g   = (w1 > v1) || (w1 == v1 && j1 < i1);
      float nv1 = g ? w1 : v1; int ni1 = g ? j1 : i1;
      float ls  = g ? v1 : w1; int li  = g ? i1 : j1;  // loser of the top compare
      float ws  = g ? w2 : v2; int wi  = g ? j2 : i2;  // winner's own second
      bool  s   = (ls > ws) || (ls == ws && li < wi);
      v2 = s ? ls : ws; i2 = s ? li : wi;
      v1 = nv1; i1 = ni1;
    }

    // 2-element masked softmax in closed form.
    float t   = __expf(v2 - v1);
    float inv = 1.0f / (1.0f + t);
    float p1  = inv;
    float p2  = t * inv;

#pragma unroll
    for (int nt = 0; nt < 4; ++nt) {
      const int e = nt * 16 + lid;
      float o = (e == i1) ? p1 : ((e == i2) ? p2 : 0.0f);
      __builtin_nontemporal_store(o, out + (size_t)row * E_DIM + e);
    }
    if (lid == 0) {
      idx_out[(size_t)row * 2 + 0] = i1;
      idx_out[(size_t)row * 2 + 1] = i2;
    }
  }
}

extern "C" void kernel_launch(void* const* d_in, const int* in_sizes, int n_in,
                              void* d_out, int out_size, void* d_ws, size_t ws_size,
                              hipStream_t stream) {
  (void)in_sizes; (void)n_in; (void)out_size; (void)ws_size;

  const float* mh = (const float*)d_in[0];  // [B,T,D]
  const float* Wr = (const float*)d_in[1];  // [D,E]
  const float* br = (const float*)d_in[2];  // [E]
  const float* Wn = (const float*)d_in[3];  // [D,E]
  const float* bn = (const float*)d_in[4];  // [E]
  const float* nu = (const float*)d_in[5];  // [B,T,E]
  // d_in[6] = top_k (compile-time K=2 here)

  uint16_t* WtC = (uint16_t*)d_ws;  // combined bf16 weights, 256KB

  float* out     = (float*)d_out;                            // [N,E] probs
  int*   idx_out = (int*)(out + (size_t)N_ROWS * E_DIM);     // [N,2] indices

  hipLaunchKernelGGL(_router_prep_weights, dim3(E_DIM), dim3(256), 0, stream,
                     Wr, Wn, WtC);
  hipLaunchKernelGGL(_router_main, dim3(N_ROWS / 64), dim3(128), 0, stream,
                     mh, br, bn, nu, WtC, out, idx_out);
}
